// ButterflyLinear_25769803815
// MI455X (gfx1250) — matrix operand order repaired and checked by reading, not compiled, not run on hardware
//
#include <hip/hip_runtime.h>
#include <hip/hip_bf16.h>

// ButterflyLinear: y = x @ (mask*W)^T + b
// x: (16384, 1024) f32, W/mask: (4096, 1024) f32, b: (4096,), y: (16384, 4096) f32
// Strategy: split f32 into bf16 hi/lo, 3x V_WMMA_F32_16X16X32_BF16 per K=32 step.
// Staging: B tiles via GLOBAL_LOAD_ASYNC_TO_LDS_B128 (ASYNCcnt), double-buffered LDS.

#define IN_F   1024
#define OUT_F  4096
#define MROWS  16384

#define BM 128
#define BN 128
#define BK 32
#define KPAD 40          // padded LDS row (bf16 elems) to spread banks
#define TPB 256          // 8 wave32s

typedef __attribute__((ext_vector_type(16))) __bf16 v16bf;
typedef __attribute__((ext_vector_type(8)))  __bf16 v8bf;
typedef __attribute__((ext_vector_type(4)))  __bf16 v4bf;
typedef __attribute__((ext_vector_type(8)))  float  v8f;

#ifndef __has_builtin
#define __has_builtin(x) 0
#endif
#if __has_builtin(__builtin_amdgcn_global_load_async_to_lds_b128)
#define HAVE_ASYNC_LDS 1
#else
#define HAVE_ASYNC_LDS 0
#endif

// 16-byte global -> LDS copy: async (no VGPR data round-trip) when available.
// Builtin signature (from hipcc diagnostic): param0 is int4-vector pointer in
// the global address space; LDS side in addrspace(3).
__device__ __forceinline__ void copy16_g2l(const void* gsrc, void* ldst) {
#if HAVE_ASYNC_LDS
  typedef int v4i __attribute__((vector_size(16)));
  typedef __attribute__((address_space(1))) v4i gv4i;
  typedef __attribute__((address_space(3))) v4i lv4i;
  __builtin_amdgcn_global_load_async_to_lds_b128(
      (gv4i*)gsrc, (lv4i*)ldst, 0, 0);
#else
  *(uint4*)ldst = *(const uint4*)gsrc;
#endif
}

__device__ __forceinline__ void wait_async0() {
#if HAVE_ASYNC_LDS
#if __has_builtin(__builtin_amdgcn_s_wait_asynccnt)
  __builtin_amdgcn_s_wait_asynccnt(0);
#else
  asm volatile("s_wait_asynccnt 0x0" ::: "memory");
#endif
#endif
}

// ---------------------------------------------------------------------------
// Stage 1: Wm = mask * W, split into bf16 hi/lo planes (row-major (OUT_F, IN_F))
// ---------------------------------------------------------------------------
__global__ __launch_bounds__(256) void bfly_prep_w(
    const float* __restrict__ W, const float* __restrict__ M,
    unsigned short* __restrict__ Wh, unsigned short* __restrict__ Wl) {
  const size_t i = ((size_t)blockIdx.x * 256 + threadIdx.x) * 4;
  const float4 w = *(const float4*)(W + i);
  const float4 m = *(const float4*)(M + i);
  float p[4] = {w.x * m.x, w.y * m.y, w.z * m.z, w.w * m.w};
  v4bf h, l;
#pragma unroll
  for (int j = 0; j < 4; ++j) {
    __bf16 hj = (__bf16)p[j];
    h[j] = hj;
    l[j] = (__bf16)(p[j] - (float)hj);
  }
  *(v4bf*)(Wh + i) = h;
  *(v4bf*)(Wl + i) = l;
}

// ---------------------------------------------------------------------------
// Stage 2: tiled GEMM with bf16x3 WMMA. Workgroup: 128x128 tile, 8 waves.
// Wave (wm,wn) owns a 64x32 sub-tile = 4x2 WMMA 16x16 accumulators.
// Double-buffered LDS; B staged via async-to-LDS.
// ---------------------------------------------------------------------------
__global__ __launch_bounds__(TPB) void bfly_gemm(
    const float* __restrict__ X,
    const unsigned short* __restrict__ Whg,
    const unsigned short* __restrict__ Wlg,
    const float* __restrict__ bias,
    float* __restrict__ Y) {
  // [buf][plane: 0=hi 1=lo][row][k]
  __shared__ __align__(16) __bf16 As[2][2][BM][KPAD];
  __shared__ __align__(16) __bf16 Bs[2][2][BN][KPAD];

  const int tid  = threadIdx.x;
  const int lane = tid & 31;
  const int wave = tid >> 5;
  const int wm   = wave >> 2;   // 0..1 -> 64-row band in M
  const int wn   = wave & 3;    // 0..3 -> 32-col band in N

  const int m0 = blockIdx.y * BM;
  const int n0 = blockIdx.x * BN;

  const int lm = lane & 15;     // M (A) / N (B) index within 16
  const int kg = lane >> 4;     // K half-select per ISA layout

  v8f acc[4][2];
#pragma unroll
  for (int ms = 0; ms < 4; ++ms)
#pragma unroll
    for (int ns = 0; ns < 2; ++ns)
      acc[ms][ns] = (v8f)(0.0f);

  // --- B staging: pure bf16 copy -> async global->LDS, no VGPR data movement
  auto stage_b = [&](int buf, int k0) {
#pragma unroll
    for (int i = 0; i < 4; ++i) {
      const int idx = tid + i * TPB;        // 0..1023 (plane uniform per i)
      const int pl  = idx >> 9;             // 0..1
      const int r   = (idx >> 2) & 127;     // 0..127
      const int c   = (idx & 3) << 3;       // 0,8,16,24 (bf16 elems; 16B)
      const unsigned short* src =
          (pl ? Wlg : Whg) + (size_t)(n0 + r) * IN_F + k0 + c;
      copy16_g2l(src, &Bs[buf][pl][r][c]);
    }
  };

  // --- A staging: f32 -> (hi,lo) bf16 conversion must pass through VGPRs
  auto stage_a = [&](int buf, int k0) {
#pragma unroll
    for (int i = 0; i < 4; ++i) {
      const int idx = tid + i * TPB;        // 0..1023
      const int r   = idx >> 3;             // 0..127
      const int c   = (idx & 7) << 2;       // 0,4,...,28
      const float4 v = *(const float4*)(X + (size_t)(m0 + r) * IN_F + k0 + c);
      float q[4] = {v.x, v.y, v.z, v.w};
      v4bf h, l;
#pragma unroll
      for (int j = 0; j < 4; ++j) {
        __bf16 hj = (__bf16)q[j];
        h[j] = hj;
        l[j] = (__bf16)(q[j] - (float)hj);
      }
      *(v4bf*)&As[buf][0][r][c] = h;
      *(v4bf*)&As[buf][1][r][c] = l;
    }
    // prefetch the X tile two k-steps ahead (X is the HBM-resident stream)
    if (k0 + 2 * BK < IN_F) {
      __builtin_prefetch(X + (size_t)(m0 + (tid >> 1)) * IN_F + k0 + 2 * BK, 0, 1);
    }
  };

  // --- fragment loads per ISA 7.12.2 (wave32):
  // A 16x32: lane<16 holds K{0..7,16..23}, lane>=16 holds K{8..15,24..31}
  auto ld_a = [&](int buf, int pl, int mrow) -> v16bf {
    const v8bf a0 = *(const v8bf*)&As[buf][pl][mrow][kg * 8];
    const v8bf a1 = *(const v8bf*)&As[buf][pl][mrow][16 + kg * 8];
    return __builtin_shufflevector(a0, a1, 0, 1, 2, 3, 4, 5, 6, 7,
                                   8, 9, 10, 11, 12, 13, 14, 15);
  };
  // B 32x16: lane = N (+16 for K-half), lane<16 holds K{0..7,16..23}, lane>=16 holds K{8..15,24..31}
  auto ld_b = [&](int buf, int pl, int nrow) -> v16bf {
    const v8bf b0 = *(const v8bf*)&Bs[buf][pl][nrow][kg * 8];
    const v8bf b1 = *(const v8bf*)&Bs[buf][pl][nrow][16 + kg * 8];
    return __builtin_shufflevector(b0, b1, 0, 1, 2, 3, 4, 5, 6, 7,
                                   8, 9, 10, 11, 12, 13, 14, 15);
  };

  // --- pipelined main loop: stage buf^1 while computing buf
  stage_b(0, 0);
  stage_a(0, 0);
  wait_async0();
  __syncthreads();

  int buf = 0;
  const int NK = IN_F / BK;  // 32 k-steps
  for (int kk = 0; kk < NK; ++kk) {
    if (kk + 1 < NK) {
      stage_b(buf ^ 1, (kk + 1) * BK);
      stage_a(buf ^ 1, (kk + 1) * BK);
    }

    v16bf ah[4], al[4], bh[2], bl[2];
#pragma unroll
    for (int ms = 0; ms < 4; ++ms) {
      const int mrow = wm * 64 + ms * 16 + lm;
      ah[ms] = ld_a(buf, 0, mrow);
      al[ms] = ld_a(buf, 1, mrow);
    }
#pragma unroll
    for (int ns = 0; ns < 2; ++ns) {
      const int nrow = wn * 32 + ns * 16 + lm;
      bh[ns] = ld_b(buf, 0, nrow);
      bl[ns] = ld_b(buf, 1, nrow);
    }

#pragma unroll
    for (int ms = 0; ms < 4; ++ms)
#pragma unroll
      for (int ns = 0; ns < 2; ++ns) {
        acc[ms][ns] = __builtin_amdgcn_wmma_f32_16x16x32_bf16(
            false, ah[ms], false, bh[ns], (short)0, acc[ms][ns], false, false);
        acc[ms][ns] = __builtin_amdgcn_wmma_f32_16x16x32_bf16(
            false, ah[ms], false, bl[ns], (short)0, acc[ms][ns], false, false);
        acc[ms][ns] = __builtin_amdgcn_wmma_f32_16x16x32_bf16(
            false, al[ms], false, bh[ns], (short)0, acc[ms][ns], false, false);
      }

    wait_async0();   // own async B copies into buf^1 complete
    __syncthreads(); // all waves' stores visible; safe to swap buffers
    buf ^= 1;
  }

  // --- epilogue: C/D layout: VGPR r -> M=r (lanes 0-15) / M=r+8 (lanes 16-31)
  const int mh = (lane >> 4) << 3;  // 0 or 8
#pragma unroll
  for (int ms = 0; ms < 4; ++ms) {
#pragma unroll
    for (int ns = 0; ns < 2; ++ns) {
      const int n  = n0 + wn * 32 + ns * 16 + lm;
      const float bv = bias[n];
      const int mb = m0 + wm * 64 + ms * 16 + mh;
#pragma unroll
      for (int r = 0; r < 8; ++r) {
        Y[(size_t)(mb + r) * OUT_F + n] = acc[ms][ns][r] + bv;
      }
    }
  }
}

// ---------------------------------------------------------------------------
extern "C" void kernel_launch(void* const* d_in, const int* in_sizes, int n_in,
                              void* d_out, int out_size, void* d_ws, size_t ws_size,
                              hipStream_t stream) {
  (void)in_sizes; (void)n_in; (void)out_size; (void)ws_size;
  const float* x    = (const float*)d_in[0];
  const float* w    = (const float*)d_in[1];
  const float* bias = (const float*)d_in[2];
  const float* mask = (const float*)d_in[3];
  float* y = (float*)d_out;

  // ws: bf16 hi plane then lo plane of masked weight (16 MB total)
  unsigned short* Wh = (unsigned short*)d_ws;
  unsigned short* Wl = Wh + (size_t)OUT_F * IN_F;

  bfly_prep_w<<<(OUT_F * IN_F) / (256 * 4), 256, 0, stream>>>(w, mask, Wh, Wl);
  bfly_gemm<<<dim3(OUT_F / BN, MROWS / BM), TPB, 0, stream>>>(x, Wh, Wl, bias, y);
}
